// GenPhiloText_9302899163623
// MI455X (gfx1250) — matrix operand order repaired and checked by reading, not compiled, not run on hardware
//
#include <hip/hip_runtime.h>
#include <hip/hip_bf16.h>

// ---------------------------------------------------------------------------
// GenPhiloText on gfx1250: embed -> (xs@W precomputed) -> 128x LSTM steps
// (h@U via WMMA, all 4 gates per wave in registers) -> projection GEMM
// (hs@Wd) with folded BatchNorm epilogue.
// GEMM core: each wave computes a 16x64 strip (4 accumulators, shared A-frag).
// Big GEMMs stage shared B tiles through LDS with double-buffered
// GLOBAL_LOAD_ASYNC_TO_LDS_B128 (ASYNCcnt) when the toolchain exposes it.
// ---------------------------------------------------------------------------

typedef __bf16 bf16_t;
typedef __attribute__((ext_vector_type(16))) __bf16 v16bf;
typedef __attribute__((ext_vector_type(8)))  __bf16 v8bf;
typedef __attribute__((ext_vector_type(8)))  float  v8f;
typedef __attribute__((ext_vector_type(4)))  int    v4i;

#define BB   32
#define TT   128
#define VV   32000
#define EE   512
#define HH   1024
#define NG   4096            // 4*H gate width
#define MROWS (TT*BB)        // 4096

#if defined(__has_builtin)
#if __has_builtin(__builtin_amdgcn_global_load_async_to_lds_b128) && \
    __has_builtin(__builtin_amdgcn_s_wait_asynccnt)
#define USE_ASYNC_LDS 1
#endif
#endif
#ifndef USE_ASYNC_LDS
#define USE_ASYNC_LDS 0
#endif

#if USE_ASYNC_LDS
typedef __attribute__((address_space(1))) v4i as1_v4i;   // global int4
typedef __attribute__((address_space(3))) v4i as3_v4i;   // LDS int4
// one 16-byte lane-granular async copy, global -> LDS (ASYNCcnt-tracked)
__device__ __forceinline__ void asyncCopy16(const bf16_t* g, bf16_t* l) {
  __builtin_amdgcn_global_load_async_to_lds_b128(
      (as1_v4i*)(unsigned long long)g,
      (as3_v4i*)(unsigned int)(unsigned long long)l, 0, 0);
}
#endif

// ---------------------------------------------------------------------------
// Packed B layout for [K,N] (K mult of 32, N mult of 16), matching the CDNA5
// 16-bit B-matrix fragment: tile (nb,kb) holds 512 bf16; lane l<16 = column
// nb*16+l with K = kb*32 + {0..7, 16..23} ; lane l>=16 same column, K = +{8..15, 24..31}.
// flat index = ((nb*KT + kb)*32 + lane)*16 + j
// ---------------------------------------------------------------------------
__global__ void pack_b_kernel(const float* __restrict__ src, bf16_t* __restrict__ dst,
                              int K, int N) {
  int gid = blockIdx.x * 256 + threadIdx.x;
  if (gid >= K * N) return;
  int k = gid / N, n = gid - k * N;
  int kb = k >> 5, kin = k & 31;
  int nb = n >> 4, nin = n & 15;
  int lane = (((kin >> 3) & 1) << 4) | nin;
  int j = (kin & 7) | ((kin >> 4) << 3);
  int KT = K >> 5;
  dst[((size_t)((nb * KT + kb) * 32 + lane) << 4) + j] = (bf16_t)src[(size_t)k * N + n];
}

// A-fragment for 16-bit 16x32 A: lane l<16 holds row l, K {0..7, 16..23};
// lane l>=16 holds row l-16, K {8..15, 24..31}.
__device__ __forceinline__ v16bf loadA(const bf16_t* __restrict__ arow) {
  union { v16bf v; v8bf h[2]; } a;
  a.h[0] = *(const v8bf*)(arow);
  a.h[1] = *(const v8bf*)(arow + 16);
  return a.v;
}

// ---------------------------------------------------------------------------
// Direct-load variant (used by the LSTM step, where waves do NOT share B):
// 16(M) x 64(N) strip per wave, double-buffered register fragments.
// ---------------------------------------------------------------------------
__device__ __forceinline__ void gemm4(const bf16_t* __restrict__ A, int lda,
                                      const bf16_t* __restrict__ Bp, int kTiles,
                                      int nbBase, int nbStride, int rowBase,
                                      v8f acc[4]) {
  const int lane = threadIdx.x & 31;
  const bf16_t* arow = A + (size_t)(rowBase + (lane & 15)) * lda + ((lane >> 4) << 3);
  const bf16_t* bp[4];
#pragma unroll
  for (int j = 0; j < 4; ++j)
    bp[j] = Bp + ((size_t)(nbBase + j * nbStride) * kTiles << 9) + (lane << 4);

  v16bf aCur = loadA(arow);
  v16bf bCur[4], bNxt[4];
#pragma unroll
  for (int j = 0; j < 4; ++j) bCur[j] = *(const v16bf*)bp[j];

#pragma unroll 2
  for (int kb = 0; kb < kTiles - 1; ++kb) {
    v16bf aNxt = loadA(arow + (kb + 1) * 32);
#pragma unroll
    for (int j = 0; j < 4; ++j)
      bNxt[j] = *(const v16bf*)(bp[j] + ((size_t)(kb + 1) << 9));
    __builtin_prefetch((const void*)(bp[0] + ((size_t)(kb + 3) << 9)), 0, 1);
#pragma unroll
    for (int j = 0; j < 4; ++j)
      acc[j] = __builtin_amdgcn_wmma_f32_16x16x32_bf16(false, aCur, false, bCur[j],
                                                       (short)0, acc[j], false, false);
    aCur = aNxt;
#pragma unroll
    for (int j = 0; j < 4; ++j) bCur[j] = bNxt[j];
  }
#pragma unroll
  for (int j = 0; j < 4; ++j)
    acc[j] = __builtin_amdgcn_wmma_f32_16x16x32_bf16(false, aCur, false, bCur[j],
                                                     (short)0, acc[j], false, false);
}

// ---------------------------------------------------------------------------
// LDS-staged variant (big GEMMs): all 8 waves of the WG share the same 4 B
// tiles per K-step, so stage them once into LDS (async copies, double
// buffered) instead of 8x redundant VMEM reads.
// smem: 2 buffers x 4 tiles x 512 bf16 = 8KB.
// ---------------------------------------------------------------------------
__device__ __forceinline__ void stageB(const bf16_t* __restrict__ Bp, int kTiles,
                                       int nbBase, int kb, bf16_t* __restrict__ dstBuf) {
  const int tid = threadIdx.x;          // 256 threads: 64 per tile, 16B each
  const int j = tid >> 6;
  const int e = (tid & 63) << 3;        // element offset (8 bf16 = 16B)
  const bf16_t* src = Bp + (((size_t)(nbBase + j) * kTiles + kb) << 9) + e;
  bf16_t* dst = dstBuf + (j << 9) + e;
#if USE_ASYNC_LDS
  asyncCopy16(src, dst);
#else
  *(v8bf*)dst = *(const v8bf*)src;
#endif
  if (kb + 4 < kTiles)
    __builtin_prefetch((const void*)(src + 2048), 0, 1);   // +4 K-steps ahead
}

__device__ __forceinline__ void gemm4_lds(const bf16_t* __restrict__ A, int lda,
                                          const bf16_t* __restrict__ Bp, int kTiles,
                                          int nbBase, int rowBase,
                                          bf16_t* __restrict__ smem /*2*4*512*/,
                                          v8f acc[4]) {
  const int lane = threadIdx.x & 31;
  const bf16_t* arow = A + (size_t)(rowBase + (lane & 15)) * lda + ((lane >> 4) << 3);

  stageB(Bp, kTiles, nbBase, 0, smem);
  for (int kb = 0; kb < kTiles - 1; ++kb) {
    bf16_t* cur = smem + ((kb & 1) << 11);
    bf16_t* nxt = smem + (((kb + 1) & 1) << 11);
    stageB(Bp, kTiles, nbBase, kb + 1, nxt);      // overlaps with this step's WMMAs
    v16bf aF = loadA(arow + kb * 32);
#if USE_ASYNC_LDS
    __builtin_amdgcn_s_wait_asynccnt(1);          // group kb landed; kb+1 in flight
#endif
    __syncthreads();                              // all waves see buffer `cur`
    v16bf bF[4];
#pragma unroll
    for (int j = 0; j < 4; ++j)
      bF[j] = *(const v16bf*)(cur + (j << 9) + (lane << 4));
#pragma unroll
    for (int j = 0; j < 4; ++j)
      acc[j] = __builtin_amdgcn_wmma_f32_16x16x32_bf16(false, aF, false, bF[j],
                                                       (short)0, acc[j], false, false);
    __syncthreads();                              // `cur` free for overwrite
  }
  {
    const int kb = kTiles - 1;
    bf16_t* cur = smem + ((kb & 1) << 11);
    v16bf aF = loadA(arow + kb * 32);
#if USE_ASYNC_LDS
    __builtin_amdgcn_s_wait_asynccnt(0);
#endif
    __syncthreads();
    v16bf bF[4];
#pragma unroll
    for (int j = 0; j < 4; ++j)
      bF[j] = *(const v16bf*)(cur + (j << 9) + (lane << 4));
#pragma unroll
    for (int j = 0; j < 4; ++j)
      acc[j] = __builtin_amdgcn_wmma_f32_16x16x32_bf16(false, aF, false, bF[j],
                                                       (short)0, acc[j], false, false);
  }
}

// ---------------------------------------------------------------------------
// Embedding: xs[r,e] = bf16(emb[X[b,t], e]),  r = t*B + b (time-major rows)
// ---------------------------------------------------------------------------
__global__ void embed_kernel(const int* __restrict__ X, const float* __restrict__ emb,
                             bf16_t* __restrict__ xs) {
  int gid = blockIdx.x * 256 + threadIdx.x;          // MROWS*EE = 2M
  if (gid >= MROWS * EE) return;
  int r = gid >> 9, e = gid & (EE - 1);
  int t = r >> 5, b = r & (BB - 1);
  int tok = X[b * TT + t];
  xs[gid] = (bf16_t)emb[(size_t)tok * EE + e];
}

__global__ void init_state_kernel(const float* __restrict__ h0, const float* __restrict__ c0,
                                  bf16_t* __restrict__ hp, float* __restrict__ cst) {
  int gid = blockIdx.x * 256 + threadIdx.x;          // 32*1024
  if (gid >= BB * HH) return;
  hp[gid]  = (bf16_t)h0[gid];
  cst[gid] = c0[gid];
}

// Fold BatchNorm (inference) + decoder bias into per-column scale/shift:
// out = acc*scale + shift ; scale = gamma*rsqrt(var+eps),
// shift = beta + (bd - mean)*scale
__global__ void bn_prep_kernel(const float* __restrict__ gamma, const float* __restrict__ beta,
                               const float* __restrict__ mean,  const float* __restrict__ var,
                               const float* __restrict__ bd,
                               float* __restrict__ scale, float* __restrict__ shift) {
  int v = blockIdx.x * 256 + threadIdx.x;
  if (v >= VV) return;
  float s = gamma[v] * rsqrtf(var[v] + 1e-3f);
  scale[v] = s;
  shift[v] = beta[v] + (bd[v] - mean[v]) * s;
}

// ---------------------------------------------------------------------------
// ZX = xs @ W + b   [4096 x 4096]   WG = 128(M) x 64(N), wave w owns M rows
// w*16..w*16+15 across 4 shared N tiles (LDS-staged).
// ---------------------------------------------------------------------------
__global__ void gemm_zx_kernel(const bf16_t* __restrict__ xs, const bf16_t* __restrict__ Wp,
                               const float* __restrict__ bias, float* __restrict__ ZX) {
  __shared__ bf16_t smem[2 * 4 * 512];
  const int wave = threadIdx.x >> 5;
  const int lane = threadIdx.x & 31;
  const int rowBase = blockIdx.y * 128 + wave * 16;
  const int nbBase  = blockIdx.x * 4;
  v8f acc[4] = {};
  gemm4_lds(xs, EE, Wp, EE / 32, nbBase, rowBase, smem, acc);
  const int r0 = rowBase + ((lane & 16) >> 1);       // +8 for upper half-wave
#pragma unroll
  for (int j = 0; j < 4; ++j) {
    const int col = (nbBase + j) * 16 + (lane & 15);
    const float bv = bias[col];
#pragma unroll
    for (int r = 0; r < 8; ++r)
      ZX[(size_t)(r0 + r) * NG + col] = acc[j][r] + bv;
  }
}

// ---------------------------------------------------------------------------
// One LSTM timestep, fully register-resident gates. grid = (H/16)/4 = 16 WGs;
// wave w: mt = w&1 (M tile of 32 rows), q = w>>1 (hidden 16-col block within
// the WG's 64-col slice). The 4 accumulators are gates i,f,g,o for the SAME
// (m, hcol) tile: B tiles strided by H/16 in the 4096-wide gate matrix.
// No LDS, no barrier; c/h update straight from accumulators.
// ---------------------------------------------------------------------------
__device__ __forceinline__ float sigm(float x) { return 1.0f / (1.0f + __expf(-x)); }

__global__ void lstm_step_kernel(const float* __restrict__ ZXt,   // [32, 4096] slice
                                 const bf16_t* __restrict__ Up,   // packed [1024,4096]
                                 const bf16_t* __restrict__ hprev,// [32,1024]
                                 bf16_t* __restrict__ hnext,      // [32,1024]
                                 float*  __restrict__ cst,        // [32,1024]
                                 bf16_t* __restrict__ hs_t) {     // hs slice [32,1024]
  const int wave = threadIdx.x >> 5;
  const int lane = threadIdx.x & 31;
  const int mt = wave & 1;
  const int q  = wave >> 1;                          // 0..3
  const int hb = blockIdx.x * 4 + q;                 // hidden 16-col block, 0..63
  v8f acc[4] = {};
  // nb(j) = j*(H/16) + hb  -> gate j's tile for this hidden block
  gemm4(hprev, HH, Up, HH / 32, hb, HH / 16, mt * 16, acc);

  const int hcol = hb * 16 + (lane & 15);
  const int mrow = mt * 16 + ((lane & 16) >> 1);
#pragma unroll
  for (int r = 0; r < 8; ++r) {
    const int m = mrow + r;
    const float* zr = ZXt + (size_t)m * NG + hcol;
    const float iv = acc[0][r] + zr[0 * HH];
    const float fv = acc[1][r] + zr[1 * HH];
    const float gv = acc[2][r] + zr[2 * HH];
    const float ov = acc[3][r] + zr[3 * HH];
    const float cn = sigm(fv) * cst[m * HH + hcol] + sigm(iv) * tanhf(gv);
    const float hn = sigm(ov) * tanhf(cn);
    cst[m * HH + hcol] = cn;
    const bf16_t hb16 = (bf16_t)hn;
    hnext[m * HH + hcol] = hb16;
    hs_t[m * HH + hcol]  = hb16;
  }
}

// ---------------------------------------------------------------------------
// out[r, v] = (hs @ Wd)[r, v] * scale[v] + shift[v]     [4096 x 32000]
// ---------------------------------------------------------------------------
__global__ void gemm_proj_kernel(const bf16_t* __restrict__ hs, const bf16_t* __restrict__ Wdp,
                                 const float* __restrict__ scale, const float* __restrict__ shift,
                                 float* __restrict__ out) {
  __shared__ bf16_t smem[2 * 4 * 512];
  const int wave = threadIdx.x >> 5;
  const int lane = threadIdx.x & 31;
  const int rowBase = blockIdx.y * 128 + wave * 16;
  const int nbBase  = blockIdx.x * 4;
  v8f acc[4] = {};
  gemm4_lds(hs, HH, Wdp, HH / 32, nbBase, rowBase, smem, acc);
  const int r0 = rowBase + ((lane & 16) >> 1);
#pragma unroll
  for (int j = 0; j < 4; ++j) {
    const int col = (nbBase + j) * 16 + (lane & 15);
    const float s = scale[col], sh = shift[col];
#pragma unroll
    for (int r = 0; r < 8; ++r)
      out[(size_t)(r0 + r) * VV + col] = acc[j][r] * s + sh;
  }
}

// ---------------------------------------------------------------------------
extern "C" void kernel_launch(void* const* d_in, const int* in_sizes, int n_in,
                              void* d_out, int out_size, void* d_ws, size_t ws_size,
                              hipStream_t stream) {
  const int*   X     = (const int*)  d_in[0];
  const float* h0    = (const float*)d_in[1];
  const float* c0    = (const float*)d_in[2];
  const float* emb   = (const float*)d_in[3];
  const float* W     = (const float*)d_in[4];
  const float* U     = (const float*)d_in[5];
  const float* bgate = (const float*)d_in[6];
  const float* Wd    = (const float*)d_in[7];
  const float* bd    = (const float*)d_in[8];
  const float* gamma = (const float*)d_in[9];
  const float* beta  = (const float*)d_in[10];
  const float* mmean = (const float*)d_in[11];
  const float* mvar  = (const float*)d_in[12];
  float* out = (float*)d_out;

  // -------- workspace carve-up (256B aligned) --------
  char* base = (char*)d_ws;
  size_t off = 0;
  auto carve = [&](size_t bytes) -> char* {
    char* p = base + off;
    off = (off + bytes + 255) & ~(size_t)255;
    return p;
  };
  bf16_t* Wp    = (bf16_t*)carve((size_t)EE * NG * 2);
  bf16_t* Up    = (bf16_t*)carve((size_t)HH * NG * 2);
  bf16_t* Wdp   = (bf16_t*)carve((size_t)HH * VV * 2);
  bf16_t* xs    = (bf16_t*)carve((size_t)MROWS * EE * 2);
  float*  ZX    = (float*) carve((size_t)MROWS * NG * 4);
  bf16_t* hsAll = (bf16_t*)carve((size_t)MROWS * HH * 2);
  bf16_t* hping = (bf16_t*)carve((size_t)BB * HH * 2);
  bf16_t* hpong = (bf16_t*)carve((size_t)BB * HH * 2);
  float*  cst   = (float*) carve((size_t)BB * HH * 4);
  float*  cScal = (float*) carve((size_t)VV * 4);
  float*  cShft = (float*) carve((size_t)VV * 4);
  (void)ws_size; (void)in_sizes; (void)n_in; (void)out_size;

  // -------- stage 0: pack weights / prep --------
  pack_b_kernel<<<(EE * NG + 255) / 256, 256, 0, stream>>>(W,  Wp,  EE, NG);
  pack_b_kernel<<<(HH * NG + 255) / 256, 256, 0, stream>>>(U,  Up,  HH, NG);
  pack_b_kernel<<<((int)((size_t)HH * VV) + 255) / 256, 256, 0, stream>>>(Wd, Wdp, HH, VV);
  embed_kernel<<<(MROWS * EE + 255) / 256, 256, 0, stream>>>(X, emb, xs);
  init_state_kernel<<<(BB * HH + 255) / 256, 256, 0, stream>>>(h0, c0, hping, cst);
  bn_prep_kernel<<<(VV + 255) / 256, 256, 0, stream>>>(gamma, beta, mmean, mvar, bd,
                                                       cScal, cShft);

  // -------- stage 1: ZX = xs @ W + b --------
  gemm_zx_kernel<<<dim3(NG / 64, MROWS / 128), 256, 0, stream>>>(xs, Wp, bgate, ZX);

  // -------- stage 2: sequential LSTM scan (h ping-ponged across launches) ----
  bf16_t* hin = hping;
  bf16_t* hout = hpong;
  for (int t = 0; t < TT; ++t) {
    lstm_step_kernel<<<(HH / 16) / 4, 256, 0, stream>>>(
        ZX + (size_t)t * BB * NG, Up, hin, hout, cst,
        hsAll + (size_t)t * BB * HH);
    bf16_t* tmp = hin; hin = hout; hout = tmp;
  }

  // -------- stage 3: projection + folded BatchNorm --------
  gemm_proj_kernel<<<dim3(VV / 64, MROWS / 128), 256, 0, stream>>>(
      hsAll, Wdp, cScal, cShft, out);
}
